// GATv2_37477884625103
// MI455X (gfx1250) — matrix operand order repaired and checked by reading, not compiled, not run on hardware
//
#include <hip/hip_runtime.h>

#define NEG_SLOPE 0.2f

typedef _Float16 v16h __attribute__((ext_vector_type(16)));
typedef _Float16 v8h  __attribute__((ext_vector_type(8)));
typedef float    v8f  __attribute__((ext_vector_type(8)));

// ---------- order-preserving float<->uint encoding for atomicMax ----------
__device__ __forceinline__ unsigned encf(float x) {
  unsigned b = __float_as_uint(x);
  return (b & 0x80000000u) ? ~b : (b | 0x80000000u);
}
__device__ __forceinline__ float decf(unsigned u) {
  return __uint_as_float((u & 0x80000000u) ? (u & 0x7FFFFFFFu) : ~u);
}

// ---------- prep kernels ----------
__global__ void k_deg(const int* __restrict__ ei, const float* __restrict__ ew,
                      float* __restrict__ cnt, float* __restrict__ wsum, int E) {
  int i = blockIdx.x * blockDim.x + threadIdx.x;
  if (i >= E) return;
  int d = ei[E + i];
  atomicAdd(&cnt[d], 1.0f);
  atomicAdd(&wsum[d], ew[i]);
}

__global__ void k_loopattr(const float* __restrict__ cnt, float* __restrict__ la, int n) {
  int i = blockIdx.x * blockDim.x + threadIdx.x;
  if (i >= n) return;
  la[i] = la[i] / fmaxf(cnt[i], 1.0f);
}

__global__ void k_tof16(const float* __restrict__ a, _Float16* __restrict__ b, int n) {
  int i = blockIdx.x * blockDim.x + threadIdx.x;
  if (i >= n) return;
  b[i] = (_Float16)a[i];
}

// W is (128 x Nout) row-major f32 -> WT is (Nout x 128) row-major f16
__global__ void k_wt(const float* __restrict__ W, _Float16* __restrict__ WT,
                     int Nout, int total) {
  int i = blockIdx.x * blockDim.x + threadIdx.x;
  if (i >= total) return;
  int k = i / Nout, n = i - k * Nout;
  WT[(size_t)n * 128 + k] = (_Float16)W[i];
}

// ---------- fused dual WMMA GEMM ----------
// Computes Cl = A@Wl + bl  and  Cr = A@Wr + br  (K fixed at 128).
// One block = 64-row strip x all Nout columns.  Wave group 0 -> Wl, group 1 -> Wr;
// within a group, one wave per 16-column tile. Each wave keeps 4 accumulators
// (one per 16-row sub-strip) so every B fragment is reused 4x.
__global__ void k_gemm2(const _Float16* __restrict__ A,
                        const _Float16* __restrict__ WTl, const _Float16* __restrict__ WTr,
                        const float* __restrict__ bl, const float* __restrict__ br,
                        float* __restrict__ Cl, float* __restrict__ Cr,
                        int nrows, int Nout) {
  constexpr int K = 128, ROWS = 64;
  __shared__ alignas(128) _Float16 sA[ROWS * K];        // 16 KB
  __shared__ alignas(128) _Float16 sW[2 * 128 * K];     // 64 KB (both weights)
  const int tid = threadIdx.x, nthr = blockDim.x;
  const int m0 = blockIdx.x * ROWS;

  // stage A strip (64 x 128 halves), zero-fill out-of-range rows
  for (int chunk = tid; chunk < ROWS * (K / 8); chunk += nthr) {
    int row = chunk / (K / 8), part = chunk - row * (K / 8);
    v8h v{};
    if (m0 + row < nrows) v = *(const v8h*)(A + (size_t)(m0 + row) * K + part * 8);
    *(v8h*)(sA + row * K + part * 8) = v;
  }
  // stage both transposed weights
  for (int chunk = tid; chunk < Nout * (K / 8); chunk += nthr) {
    *(v8h*)(sW + chunk * 8)           = *(const v8h*)(WTl + (size_t)chunk * 8);
    *(v8h*)(sW + 128 * K + chunk * 8) = *(const v8h*)(WTr + (size_t)chunk * 8);
  }
  __syncthreads();

  const int lane = tid & 31, wave = tid >> 5;
  const int ntiles = Nout >> 4;
  const int grp   = wave / ntiles;        // 0 -> Wl/Cl, 1 -> Wr/Cr
  const int ntile = wave - grp * ntiles;
  const int row   = lane & 15;            // A frag: lanes 0-15 / 16-31 hold M=0..15
  const int khalf = (lane >> 4) * 16;     // A frag: upper lane-half -> K offset +16
  const int kq    = (lane >> 4) * 16;     // B frag: upper lane-half -> K offset +16
  const int colg  = lane & 15;
  const _Float16* wb = sW + grp * (128 * K) + (size_t)(ntile * 16 + colg) * K;

  v8f c[4] = {};
  #pragma unroll
  for (int kk = 0; kk < K; kk += 32) {
    v16h b = *(const v16h*)(wb + kk + kq);
    #pragma unroll
    for (int s = 0; s < 4; s++) {
      const _Float16* ab = sA + (s * 16 + row) * K + kk + khalf;
      v8h alo = *(const v8h*)(ab);
      v8h ahi = *(const v8h*)(ab + 8);
      v16h a;
      #pragma unroll
      for (int i = 0; i < 8; i++) { a[i] = alo[i]; a[8 + i] = ahi[i]; }
      c[s] = __builtin_amdgcn_wmma_f32_16x16x32_f16(false, a, false, b,
                                                    (short)0, c[s], false, false);
    }
  }

  // epilogue: C/D layout -> VGPR r holds row r (lanes 0-15) / row r+8 (lanes 16-31)
  const int rbase = (lane >> 4) * 8;
  const int gcol  = ntile * 16 + colg;
  float* Co = grp ? Cr : Cl;
  const float bv = (grp ? br : bl)[gcol];
  if (m0 + ROWS <= nrows) {                     // uniform fast path, no guards
    #pragma unroll
    for (int s = 0; s < 4; s++) {
      size_t base = (size_t)(m0 + s * 16 + rbase) * Nout + gcol;
      #pragma unroll
      for (int r = 0; r < 8; r++) Co[base + (size_t)r * Nout] = c[s][r] + bv;
    }
  } else {
    #pragma unroll
    for (int s = 0; s < 4; s++)
      #pragma unroll
      for (int r = 0; r < 8; r++) {
        int grow = m0 + s * 16 + rbase + r;
        if (grow < nrows) Co[(size_t)grow * Nout + gcol] = c[s][r] + bv;
      }
  }
}

// ---------- edge kernels (one wave per edge) ----------
template <int F, int H>
__global__ void k_score(const float* __restrict__ xl, const float* __restrict__ xr,
                        const float* __restrict__ Wev, const float* __restrict__ att,
                        const int* __restrict__ ei, const float* __restrict__ ew,
                        const float* __restrict__ la, int E, int Etot,
                        float* __restrict__ sbuf, unsigned* __restrict__ mbuf) {
  constexpr int VPL = F / 32;   // channels per lane
  constexpr int GS  = 32 / H;   // lanes per head
  unsigned gid = blockIdx.x * blockDim.x + threadIdx.x;
  int e = (int)(gid >> 5);
  if (e >= Etot) return;
  int lane = threadIdx.x & 31;
  int s, d; float w;
  if (e < E) { s = ei[e]; d = ei[E + e]; w = ew[e]; }
  else       { s = d = e - E; w = la[e - E]; }
  int ch0 = lane * VPL;
  const float* pl = xl + (size_t)s * F + ch0;
  const float* pr = xr + (size_t)d * F + ch0;
  float acc = 0.f;
  #pragma unroll
  for (int j = 0; j < VPL; j++) {
    float t = pl[j] + pr[j] + w * Wev[ch0 + j];
    t = (t > 0.f) ? t : NEG_SLOPE * t;
    acc += t * att[ch0 + j];
  }
  #pragma unroll
  for (int off = GS >> 1; off > 0; off >>= 1) acc += __shfl_xor(acc, off, 32);
  if ((lane & (GS - 1)) == 0) {
    int h = lane / GS;
    sbuf[(size_t)e * H + h] = acc;
    atomicMax(&mbuf[(size_t)d * H + h], encf(acc));
  }
}

template <int H>
__global__ void k_exp(float* __restrict__ sbuf, const unsigned* __restrict__ mbuf,
                      float* __restrict__ denom, const int* __restrict__ ei,
                      int E, int Etot) {
  int idx = blockIdx.x * blockDim.x + threadIdx.x;
  if (idx >= Etot * H) return;
  int e = idx / H, h = idx - e * H;
  int d = (e < E) ? ei[E + e] : (e - E);
  float p = expf(sbuf[idx] - decf(mbuf[(size_t)d * H + h]));
  sbuf[idx] = p;
  atomicAdd(&denom[(size_t)d * H + h], p);
}

template <int F, int H>
__global__ void k_agg(const float* __restrict__ xl, const float* __restrict__ sbuf,
                      const float* __restrict__ denom, const int* __restrict__ ei,
                      int E, int Etot, float* __restrict__ agg) {
  constexpr int VPL = F / 32;
  constexpr int GS  = 32 / H;
  unsigned gid = blockIdx.x * blockDim.x + threadIdx.x;
  int e = (int)(gid >> 5);
  if (e >= Etot) return;
  int lane = threadIdx.x & 31;
  int s, d;
  if (e < E) { s = ei[e]; d = ei[E + e]; }
  else       { s = d = e - E; }
  int h = lane / GS;
  float alpha = sbuf[(size_t)e * H + h] / (denom[(size_t)d * H + h] + 1e-16f);
  int ch0 = lane * VPL;
  const float* pl = xl + (size_t)s * F + ch0;
  float* pa = agg + (size_t)d * F + ch0;
  #pragma unroll
  for (int j = 0; j < VPL; j++) atomicAdd(&pa[j], alpha * pl[j]);
}

// ---------- post kernels ----------
__global__ void k_post_mid(const float* __restrict__ agg, const float* __restrict__ bias,
                           _Float16* __restrict__ x16, int total, int F) {
  int i = blockIdx.x * blockDim.x + threadIdx.x;
  if (i >= total) return;
  float v = agg[i] + bias[i & (F - 1)];
  v = (v > 0.f) ? v : expm1f(v);   // ELU
  x16[i] = (_Float16)v;
}

__global__ void k_post_out(const float* __restrict__ agg, const float* __restrict__ bias,
                           float* __restrict__ out, int total, int F) {
  int i = blockIdx.x * blockDim.x + threadIdx.x;
  if (i >= total) return;
  out[i] = agg[i] + bias[i & (F - 1)];
}

// ---------- driver ----------
extern "C" void kernel_launch(void* const* d_in, const int* in_sizes, int n_in,
                              void* d_out, int out_size, void* d_ws, size_t ws_size,
                              hipStream_t stream) {
  (void)n_in; (void)out_size; (void)ws_size;
  const float* x  = (const float*)d_in[0];
  const int*   ei = (const int*)d_in[1];
  const float* ew = (const float*)d_in[2];
  const int N    = in_sizes[0] / 128;
  const int E    = in_sizes[2];
  const int Etot = E + N;
  const float* Wl[3]   = {(const float*)d_in[3],  (const float*)d_in[10], (const float*)d_in[17]};
  const float* bl[3]   = {(const float*)d_in[4],  (const float*)d_in[11], (const float*)d_in[18]};
  const float* Wr[3]   = {(const float*)d_in[5],  (const float*)d_in[12], (const float*)d_in[19]};
  const float* br[3]   = {(const float*)d_in[6],  (const float*)d_in[13], (const float*)d_in[20]};
  const float* We[3]   = {(const float*)d_in[7],  (const float*)d_in[14], (const float*)d_in[21]};
  const float* att[3]  = {(const float*)d_in[8],  (const float*)d_in[15], (const float*)d_in[22]};
  const float* bias[3] = {(const float*)d_in[9],  (const float*)d_in[16], (const float*)d_in[23]};
  const int Fo[3] = {128, 128, 64};

  char* p = (char*)d_ws;
  auto carve = [&](size_t bytes) -> void* {
    void* r = (void*)p; p += (bytes + 255) & ~(size_t)255; return r;
  };
  _Float16* x16  = (_Float16*)carve((size_t)N * 128 * 2);
  float*    xl   = (float*)carve((size_t)N * 128 * 4);
  float*    xr   = (float*)carve((size_t)N * 128 * 4);
  float*    agg  = (float*)carve((size_t)N * 128 * 4);
  float*    sbuf = (float*)carve((size_t)Etot * 4 * 4);
  unsigned* mbuf = (unsigned*)carve((size_t)N * 4 * 4);
  float*    den  = (float*)carve((size_t)N * 4 * 4);
  float*    cnt  = (float*)carve((size_t)N * 4);
  float*    la   = (float*)carve((size_t)N * 4);
  _Float16* wtl[3]; _Float16* wtr[3];
  for (int l = 0; l < 3; l++) {
    wtl[l] = (_Float16*)carve((size_t)128 * Fo[l] * 2);
    wtr[l] = (_Float16*)carve((size_t)128 * Fo[l] * 2);
  }

  auto cdiv = [](size_t a, size_t b) { return (unsigned)((a + b - 1) / b); };

  hipMemsetAsync(cnt, 0, (size_t)N * 4, stream);
  hipMemsetAsync(la,  0, (size_t)N * 4, stream);
  k_deg<<<cdiv(E, 256), 256, 0, stream>>>(ei, ew, cnt, la, E);
  k_loopattr<<<cdiv(N, 256), 256, 0, stream>>>(cnt, la, N);
  k_tof16<<<cdiv((size_t)N * 128, 256), 256, 0, stream>>>(x, x16, N * 128);
  for (int l = 0; l < 3; l++) {
    int tot = 128 * Fo[l];
    k_wt<<<cdiv(tot, 256), 256, 0, stream>>>(Wl[l], wtl[l], Fo[l], tot);
    k_wt<<<cdiv(tot, 256), 256, 0, stream>>>(Wr[l], wtr[l], Fo[l], tot);
  }

  for (int l = 0; l < 3; l++) {
    const int F = Fo[l];
    const int H = (l < 2) ? 4 : 1;
    // fused xl/xr GEMM: 64-row strips, wave group per weight matrix
    dim3 gg(cdiv(N, 64)), gb(64 * (F / 16));
    k_gemm2<<<gg, gb, 0, stream>>>(x16, wtl[l], wtr[l], bl[l], br[l], xl, xr, N, F);
    hipMemsetAsync(mbuf, 0, (size_t)N * H * 4, stream);
    hipMemsetAsync(den,  0, (size_t)N * H * 4, stream);
    hipMemsetAsync(agg,  0, (size_t)N * F * 4, stream);
    unsigned sb = cdiv((size_t)Etot * 32, 256);
    if (l < 2) {
      k_score<128, 4><<<sb, 256, 0, stream>>>(xl, xr, We[l], att[l], ei, ew, la, E, Etot, sbuf, mbuf);
      k_exp<4><<<cdiv((size_t)Etot * 4, 256), 256, 0, stream>>>(sbuf, mbuf, den, ei, E, Etot);
      k_agg<128, 4><<<sb, 256, 0, stream>>>(xl, sbuf, den, ei, E, Etot, agg);
      k_post_mid<<<cdiv((size_t)N * 128, 256), 256, 0, stream>>>(agg, bias[l], x16, N * 128, 128);
    } else {
      k_score<64, 1><<<sb, 256, 0, stream>>>(xl, xr, We[l], att[l], ei, ew, la, E, Etot, sbuf, mbuf);
      k_exp<1><<<cdiv((size_t)Etot, 256), 256, 0, stream>>>(sbuf, mbuf, den, ei, E, Etot);
      k_agg<64, 1><<<sb, 256, 0, stream>>>(xl, sbuf, den, ei, E, Etot, agg);
      k_post_out<<<cdiv((size_t)N * 64, 256), 256, 0, stream>>>(agg, bias[2], (float*)d_out, N * 64, 64);
    }
  }
}
